// RISnetPartialCSI_8495445311905
// MI455X (gfx1250) — matrix operand order repaired and checked by reading, not compiled, not run on hardware
//
#include <hip/hip_runtime.h>

typedef _Float16 half_t;
typedef __attribute__((ext_vector_type(16))) _Float16 v16h;
typedef __attribute__((ext_vector_type(8)))  _Float16 v8h;
typedef __attribute__((ext_vector_type(8)))  float    v8f;

#define THREADS 256
#define NWAVES 8

// Build A fragments (16x32 f16, M=output row, K=input channel) for a
// process layer: a0 = out channels 0..15 (branches ll,lg), a1 = 16..31 (gl,gg).
// Weight element (branch, o, c) = w[base + branch*brStride + o*cin + c].
__device__ __forceinline__ void build_A(const float* __restrict__ w, int cin,
                                        int brStride, int base, int lane,
                                        v16h& a0, v16h& a1) {
  const int Ml = lane & 15;
  const int kg = (lane >> 4) & 1;
#pragma unroll
  for (int h = 0; h < 16; ++h) {
    const int K = h + 16 * kg;   // contiguous k order, matches B fill
    float f0 = 0.f, f1 = 0.f;
    if (K < cin) {
      f0 = w[base + (Ml >> 3) * brStride + (Ml & 7) * cin + K];
      const int M1 = Ml + 16;
      f1 = w[base + (M1 >> 3) * brStride + (M1 & 7) * cin + K];
    }
    a0[h] = (_Float16)f0;
    a1[h] = (_Float16)f1;
  }
}

// Per-lane bias for D rows: D row r on this lane is out channel M = r + 8*(lane/16)
// (+16 for the second WMMA). bias element = b[bBase + branch*bStride + o].
__device__ __forceinline__ void load_bias(const float* __restrict__ b, int bStride,
                                          int bBase, int lane, float* bb0, float* bb1) {
  const int hi = (lane >> 4) & 1;
#pragma unroll
  for (int r = 0; r < 8; ++r) {
    bb0[r] = b[bBase + hi * bStride + r];
    bb1[r] = b[bBase + (2 + hi) * bStride + r];
  }
}

// One 16-pixel tile of a process layer: load B (32ch x 16px) from pixel-major
// LDS, 2x WMMA, bias+ReLU, store 32 output channels back (pixel-major).
__device__ __forceinline__ void tile_compute(const half_t* src, half_t* dst, int W,
                                             int u, int tw, int lane,
                                             const v16h& a0, const v16h& a1,
                                             const float* bb0, const float* bb1) {
  const int N = lane & 15;
  const int kg = (lane >> 4) & 1;
  const int pix = (u * W + tw * 16 + N) * 32;
  v16h bf = *(const v16h*)(src + pix + kg * 16);
  v8f c0 = {}, c1 = {};
  c0 = __builtin_amdgcn_wmma_f32_16x16x32_f16(false, a0, false, bf, (short)0, c0, false, false);
  c1 = __builtin_amdgcn_wmma_f32_16x16x32_f16(false, a1, false, bf, (short)0, c1, false, false);
  v8h o0, o1;
#pragma unroll
  for (int r = 0; r < 8; ++r) {
    o0[r] = (_Float16)fmaxf(c0[r] + bb0[r], 0.f);
    o1[r] = (_Float16)fmaxf(c1[r] + bb1[r], 0.f);
  }
  *(v8h*)(dst + pix + kg * 8) = o0;
  *(v8h*)(dst + pix + 16 + kg * 8) = o1;
}

// Replace channels 8..15 by mean over W, 16..23 by mean over U, 24..31 by mean
// over both (broadcast in place). Tasks are disjoint per channel -> no scratch.
__device__ __forceinline__ void reduce_bcast(half_t* buf, int W) {
  const int nGl = 8 * W;
  const int total = 32 + nGl + 8;
  for (int t = threadIdx.x; t < total; t += THREADS) {
    if (t < 32) {                       // lg: mean over w per (o,u)
      const int o = t >> 2, u = t & 3, c = 8 + o;
      float s = 0.f;
      for (int w = 0; w < W; ++w) s += (float)buf[(u * W + w) * 32 + c];
      const half_t m = (half_t)(s / (float)W);
      for (int w = 0; w < W; ++w) buf[(u * W + w) * 32 + c] = m;
    } else if (t < 32 + nGl) {          // gl: mean over u per (o,w)
      const int tt = t - 32, o = tt / W, w = tt % W, c = 16 + o;
      float s = 0.f;
      for (int u = 0; u < 4; ++u) s += (float)buf[(u * W + w) * 32 + c];
      const half_t m = (half_t)(s * 0.25f);
      for (int u = 0; u < 4; ++u) buf[(u * W + w) * 32 + c] = m;
    } else {                            // gg: mean over (u,w) per o
      const int o = t - 32 - nGl, c = 24 + o;
      float s = 0.f;
      for (int u = 0; u < 4; ++u)
        for (int w = 0; w < W; ++w) s += (float)buf[(u * W + w) * 32 + c];
      const half_t m = (half_t)(s / (4.f * (float)W));
      for (int u = 0; u < 4; ++u)
        for (int w = 0; w < W; ++w) buf[(u * W + w) * 32 + c] = m;
    }
  }
}

__device__ __forceinline__ void process_layer(const half_t* src, half_t* dst, int W,
    const float* w, const float* bs, int cin, int brStride, int wBase,
    int bStride, int bBase) {
  const int lane = threadIdx.x & 31;
  const int wave = threadIdx.x >> 5;
  v16h a0, a1;
  build_A(w, cin, brStride, wBase, lane, a0, a1);
  float bb0[8], bb1[8];
  load_bias(bs, bStride, bBase, lane, bb0, bb1);
  const int ntw = W >> 4;
  const int ntiles = 4 * ntw;
  for (int t = wave; t < ntiles; t += NWAVES)
    tile_compute(src, dst, W, t / ntw, t % ntw, lane, a0, a1, bb0, bb1);
  __syncthreads();
  reduce_bcast(dst, W);
  __syncthreads();
}

__global__ __launch_bounds__(THREADS) void risnet_fused(
    const float* __restrict__ channel,
    const float* __restrict__ w1, const float* __restrict__ b1,
    const float* __restrict__ w2, const float* __restrict__ b2,
    const float* __restrict__ w3, const float* __restrict__ b3,
    const float* __restrict__ w4, const float* __restrict__ b4,
    const float* __restrict__ w5, const float* __restrict__ b5,
    const float* __restrict__ w6, const float* __restrict__ b6,
    const float* __restrict__ w7, const float* __restrict__ b7,
    const float* __restrict__ w8, const float* __restrict__ b8,
    float* __restrict__ out)
{
  __shared__ __align__(16) half_t bufA[32 * 4 * 144];   // 36 KB
  __shared__ __align__(16) half_t bufB[32 * 4 * 144];   // 36 KB
  __shared__ __align__(16) half_t gtmp[32 * 4 * 16];    // 4 KB
  __shared__ float finalAcc[1296];
  __shared__ unsigned short iperm[1296];
  __shared__ float red[128];
  __shared__ float sAcc;

  const int b = blockIdx.x;
  const int tid = threadIdx.x;
  const int lane = tid & 31;
  const int wave = tid >> 5;

  // ---- phase-normalization mean over channels {1,3} x U x W (128 values) ----
  if (tid < 128) {
    const int c = 1 + 2 * (tid >> 6);
    const int u = (tid >> 4) & 3;
    const int w = tid & 15;
    red[tid] = channel[((b * 4 + c) * 4 + u) * 16 + w];
  }
  __syncthreads();
  if (tid == 0) {
    float s = 0.f;
    for (int i = 0; i < 128; ++i) s += red[i];
    sAcc = s * (1.f / 128.f);
  }
  __syncthreads();
  const float meanPhase = sAcc;

  // ---- fill bufA pixel-major [u][w][c], channels >=4 zero-padded ----
  for (int t = tid; t < 4 * 16 * 32; t += THREADS) {
    const int c = t & 31;
    const int pix = t >> 5;
    const int u = pix >> 4, w = pix & 15;
    float v = 0.f;
    if (c < 4) {
      v = channel[((b * 4 + c) * 4 + u) * 16 + w];
      if (c & 1) v -= meanPhase;
    }
    bufA[(u * 16 + w) * 32 + c] = (half_t)v;
  }
  __syncthreads();

  // ---- layers 1,2 (W=16, in place: 1x1 conv so per-pixel independent) ----
  process_layer(bufA, bufA, 16, w1, b1, 4, 32, 0, 8, 0);
  process_layer(bufA, bufA, 16, w2, b2, 32, 256, 0, 8, 0);

  // ---- inverse PERM3 ----
  for (int t = tid; t < 144; t += THREADS) {
    const int i = t / 12, j = t % 12;
    const int p = ((i % 3) * 3 + (j % 3)) * 12 + (i / 3) * 4 + (j / 3);
    iperm[p] = (unsigned short)t;
  }
  __syncthreads();

  // ---- interp layer 3: 9 process layers on f2, scattered into W=144 ----
  for (int k = 0; k < 9; ++k) {
    process_layer(bufA, gtmp, 16, w3, b3, 32, 2304, k * 256, 72, k * 8);
    for (int t = tid; t < 256; t += THREADS) {      // 64 pixels x 4 quarters
      const int q = t & 3;
      const int pix = t >> 2;
      const int u = pix >> 4, w16 = pix & 15;
      const int wp = iperm[k * 16 + w16];
      *(uint4*)(bufB + (u * 144 + wp) * 32 + q * 8) =
          *(const uint4*)(gtmp + (u * 16 + w16) * 32 + q * 8);
    }
    __syncthreads();
  }

  // ---- layers 4,5 (W=144, in place) ----
  process_layer(bufB, bufB, 144, w4, b4, 32, 256, 0, 8, 0);
  process_layer(bufB, bufB, 144, w5, b5, 32, 256, 0, 8, 0);

  // ---- inverse PERM6 + zero accumulators ----
  for (int t = tid; t < 1296; t += THREADS) {
    const int i = t / 36, j = t % 36;
    const int p = ((i % 3) * 3 + (j % 3)) * 108 + (i / 3) * 12 + (j / 3);
    iperm[p] = (unsigned short)t;
    finalAcc[t] = 0.f;
  }
  if (tid == 0) sAcc = 0.f;
  __syncthreads();

  // ---- layer7 weight fragments + per-lane w8 slices ----
  v16h a70, a71;
  build_A(w7, 32, 256, 0, lane, a70, a71);
  float b70[8], b71[8];
  load_bias(b7, 8, 0, lane, b70, b71);
  float w8a[8], w8b[8];
  {
    const int lo  = (lane < 16) ? 0 : 8;    // ll rows on lanes 0-15, lg on 16-31
    const int hi2 = (lane < 16) ? 16 : 24;  // gl rows on lanes 0-15, gg on 16-31
#pragma unroll
    for (int r = 0; r < 8; ++r) { w8a[r] = w8[lo + r]; w8b[r] = w8[hi2 + r]; }
  }

  // ---- fused interp6 (per-k) + layer7 + layer8 + mean over users ----
  // final[w'] gets sum_u (w8_ll.relu + w8_gl.relu); lg/gg collapse to one scalar
  // because their mean-broadcast makes the w8 contribution uniform over w'.
  for (int k = 0; k < 9; ++k) {
    process_layer(bufB, bufA, 144, w6, b6, 32, 2304, k * 256, 72, k * 8); // g_k
    for (int t = wave; t < 36; t += NWAVES) {
      const int u = t / 9, tw = t % 9;
      const int N = lane & 15;
      const int kg = (lane >> 4) & 1;
      v16h bf = *(const v16h*)(bufA + (u * 144 + tw * 16 + N) * 32 + kg * 16);
      v8f c0 = {}, c1 = {};
      c0 = __builtin_amdgcn_wmma_f32_16x16x32_f16(false, a70, false, bf, (short)0, c0, false, false);
      c1 = __builtin_amdgcn_wmma_f32_16x16x32_f16(false, a71, false, bf, (short)0, c1, false, false);
      float s = 0.f;
#pragma unroll
      for (int r = 0; r < 8; ++r) {
        s += w8a[r] * fmaxf(c0[r] + b70[r], 0.f);
        s += w8b[r] * fmaxf(c1[r] + b71[r], 0.f);
      }
      if (lane < 16) {                       // ll + gl: per-pixel -> final[w']
        const int wp = iperm[k * 144 + tw * 16 + N];
        atomicAdd(&finalAcc[wp], s);
      } else {                               // lg + gg: global scalar
        s += __shfl_xor(s, 1);
        s += __shfl_xor(s, 2);
        s += __shfl_xor(s, 4);
        s += __shfl_xor(s, 8);
        if (lane == 16) atomicAdd(&sAcc, s);
      }
    }
    __syncthreads();
  }

  const float b8v = b8[0];
  const float PI = 3.14159265358979323846f;
  for (int t = tid; t < 1296; t += THREADS) {
    out[b * 1296 + t] =
        PI * (finalAcc[t] * 0.25f + sAcc * (1.f / (4.f * 1296.f)) + b8v);
  }
}

extern "C" void kernel_launch(void* const* d_in, const int* in_sizes, int n_in,
                              void* d_out, int out_size, void* d_ws, size_t ws_size,
                              hipStream_t stream) {
  (void)n_in; (void)d_ws; (void)ws_size; (void)out_size;
  const float* channel = (const float*)d_in[0];
  const float* w1 = (const float*)d_in[1];
  const float* b1 = (const float*)d_in[2];
  const float* w2 = (const float*)d_in[3];
  const float* b2 = (const float*)d_in[4];
  const float* w3 = (const float*)d_in[5];
  const float* b3 = (const float*)d_in[6];
  const float* w4 = (const float*)d_in[7];
  const float* b4 = (const float*)d_in[8];
  const float* w5 = (const float*)d_in[9];
  const float* b5 = (const float*)d_in[10];
  const float* w6 = (const float*)d_in[11];
  const float* b6 = (const float*)d_in[12];
  const float* w7 = (const float*)d_in[13];
  const float* b7 = (const float*)d_in[14];
  const float* w8 = (const float*)d_in[15];
  const float* b8 = (const float*)d_in[16];
  const int batch = in_sizes[0] / (4 * 4 * 16);   // 512
  risnet_fused<<<batch, THREADS, 0, stream>>>(channel, w1, b1, w2, b2, w3, b3,
                                              w4, b4, w5, b5, w6, b6, w7, b7,
                                              w8, b8, (float*)d_out);
}
